// StarGIN_86466281603524
// MI455X (gfx1250) — compile-verified, matrix operand order checked by reading, not run on hardware
//
#include <hip/hip_runtime.h>
#include <hip/hip_bf16.h>

#define NN   100000
#define NE   1000000
#define NG   1000
#define FIN  16
#define EDIM 63
#define HD   128
#define NL   2
#define NC   36
#define BN_EPS 1e-5f

typedef __attribute__((ext_vector_type(16))) __bf16 v16bf;
typedef __attribute__((ext_vector_type(8)))  __bf16 v8bf;
typedef __attribute__((ext_vector_type(4)))  __bf16 v4bf;
typedef __attribute__((ext_vector_type(8)))  float  v8f;

static __device__ __forceinline__ v8f wmma_bf(v16bf a, v16bf b, v8f c) {
  return __builtin_amdgcn_wmma_f32_16x16x32_bf16(false, a, false, b, (short)0, c, false, false);
}

static __device__ __forceinline__ v8f vzero8() {
  v8f z;
#pragma unroll
  for (int j = 0; j < 8; ++j) z[j] = 0.0f;
  return z;
}

// Global W [K][HD] f32 row-major -> LDS wT [HD][KPAD] bf16 (transposed, zero-padded K..KPAD)
static __device__ __forceinline__ void load_weightT(const float* __restrict__ W,
                                                    __bf16* wT, int K, int KPAD) {
  for (int idx = threadIdx.x; idx < HD * KPAD; idx += blockDim.x) {
    int col = idx / KPAD, k = idx - col * KPAD;
    float v = (k < K) ? W[k * HD + col] : 0.0f;
    wT[idx] = (__bf16)v;
  }
}

// B fragment (32x16 bf16) from transposed LDS weights [HD][KPAD]
static __device__ __forceinline__ v16bf load_b_frag(const __bf16* wT, int KPAD, int nt, int k0) {
  int lane = threadIdx.x & 31;
  const __bf16* p = wT + (nt * 16 + (lane & 15)) * KPAD + k0 + ((lane & 16) ? 8 : 0);
  v8bf lo = *(const v8bf*)p;
  v8bf hi = *(const v8bf*)(p + 16);
  v16bf b;
#pragma unroll
  for (int j = 0; j < 8; ++j) { b[j] = lo[j]; b[8 + j] = hi[j]; }
  return b;
}

// A fragment (16x32 bf16) from bf16 row-major source (global or LDS)
static __device__ __forceinline__ v16bf load_a_bf16(const __bf16* A, int row0, int maxRow,
                                                    int ld, int k0) {
  int lane = threadIdx.x & 31;
  int row = row0 + (lane & 15); if (row > maxRow) row = maxRow;
  const __bf16* p = A + (long)row * ld + k0 + ((lane & 16) ? 8 : 0);
  v8bf lo = *(const v8bf*)p;
  v8bf hi = *(const v8bf*)(p + 16);
  v16bf a;
#pragma unroll
  for (int j = 0; j < 8; ++j) { a[j] = lo[j]; a[8 + j] = hi[j]; }
  return a;
}

// A fragment from f32 row-major with K-limit zero padding, converted to bf16
static __device__ __forceinline__ v16bf load_a_f32(const float* A, int row0, int maxRow,
                                                   int ld, int k0, int klim) {
  int lane = threadIdx.x & 31;
  int row = row0 + (lane & 15); if (row > maxRow) row = maxRow;
  int kb = k0 + ((lane & 16) ? 8 : 0);
  const float* p = A + (long)row * ld + kb;
  v16bf a;
#pragma unroll
  for (int j = 0; j < 8; ++j) {
    float v0 = (kb + j      < klim) ? p[j]      : 0.0f;
    float v1 = (kb + 16 + j < klim) ? p[16 + j] : 0.0f;
    a[j]     = (__bf16)v0;
    a[8 + j] = (__bf16)v1;
  }
  return a;
}

// Generic row-tiled GEMM: out[M,HD] = act(A[M,K] @ W[K,HD] + bias)
template <int K, int KPAD, bool RELU, bool OUTBF>
__global__ void gemm_kernel(const float* __restrict__ A, const float* __restrict__ W,
                            const float* __restrict__ bias, void* __restrict__ outv, int M) {
  __shared__ __align__(16) __bf16 wT[HD * KPAD];
  load_weightT(W, wT, K, KPAD);
  __syncthreads();
  int wave = threadIdx.x >> 5;
  int row0 = (blockIdx.x * (blockDim.x >> 5) + wave) << 4;
  if (row0 >= M) return;
  int lane = threadIdx.x & 31;
  int nl = lane & 15, mb = (lane & 16) ? 8 : 0;
  v8f acc[8];
#pragma unroll
  for (int t = 0; t < 8; ++t) acc[t] = vzero8();
#pragma unroll
  for (int k0 = 0; k0 < KPAD; k0 += 32) {
    v16bf a = load_a_f32(A, row0, M - 1, K, k0, K);
#pragma unroll
    for (int t = 0; t < 8; ++t)
      acc[t] = wmma_bf(a, load_b_frag(wT, KPAD, t, k0), acc[t]);
  }
  float* outf = (float*)outv;
  __bf16* outb = (__bf16*)outv;
#pragma unroll
  for (int t = 0; t < 8; ++t) {
    int col = t * 16 + nl;
    float bv = bias[col];
#pragma unroll
    for (int r = 0; r < 8; ++r) {
      int row = row0 + mb + r;
      if (row < M) {
        float v = acc[t][r] + bv;
        if (RELU) v = fmaxf(v, 0.0f);
        long off = (long)row * HD + col;
        if (OUTBF) outb[off] = (__bf16)v; else outf[off] = v;
      }
    }
  }
}

// Fused GIN MLP: Y = relu(Z @ W1 + b1) @ W2 + b2, Z is bf16 [M,HD]
__global__ void mlp_kernel(const __bf16* __restrict__ Z, const float* __restrict__ W1,
                           const float* __restrict__ B1, const float* __restrict__ W2,
                           const float* __restrict__ B2, float* __restrict__ Y, int M) {
  __shared__ __align__(16) __bf16 wT[HD * HD];        // 32 KB, reused W1 then W2
  __shared__ __align__(16) __bf16 tst[4 * 16 * HD];   // 16 KB staging, one tile per wave
  load_weightT(W1, wT, HD, HD);
  __syncthreads();
  int wave = threadIdx.x >> 5;
  int row0 = (blockIdx.x * (blockDim.x >> 5) + wave) << 4;
  bool active = (row0 < M);
  int lane = threadIdx.x & 31;
  int nl = lane & 15, mb = (lane & 16) ? 8 : 0;
  v8f acc[8];
#pragma unroll
  for (int t = 0; t < 8; ++t) acc[t] = vzero8();
  __bf16* ts = tst + wave * 16 * HD;
  if (active) {
#pragma unroll
    for (int k0 = 0; k0 < HD; k0 += 32) {
      v16bf a = load_a_bf16(Z, row0, M - 1, HD, k0);
#pragma unroll
      for (int t = 0; t < 8; ++t)
        acc[t] = wmma_bf(a, load_b_frag(wT, HD, t, k0), acc[t]);
    }
    // bias + relu, stage tile as bf16 for the second GEMM (C-frag -> A-frag transpose via LDS)
#pragma unroll
    for (int t = 0; t < 8; ++t) {
      int col = t * 16 + nl;
      float bv = B1[col];
#pragma unroll
      for (int r = 0; r < 8; ++r) {
        float v = fmaxf(acc[t][r] + bv, 0.0f);
        ts[(mb + r) * HD + col] = (__bf16)v;
      }
    }
    asm volatile("s_wait_dscnt 0x0" ::: "memory");
  }
  __syncthreads();
  load_weightT(W2, wT, HD, HD);
  __syncthreads();
  if (active) {
#pragma unroll
    for (int t = 0; t < 8; ++t) acc[t] = vzero8();
#pragma unroll
    for (int k0 = 0; k0 < HD; k0 += 32) {
      v16bf a = load_a_bf16(ts, 0, 15, HD, k0);
#pragma unroll
      for (int t = 0; t < 8; ++t)
        acc[t] = wmma_bf(a, load_b_frag(wT, HD, t, k0), acc[t]);
    }
#pragma unroll
    for (int t = 0; t < 8; ++t) {
      int col = t * 16 + nl;
      float bv = B2[col];
#pragma unroll
      for (int r = 0; r < 8; ++r) {
        int row = row0 + mb + r;
        if (row < M) Y[(long)row * HD + col] = acc[t][r] + bv;
      }
    }
  }
}

__global__ void zero_kernel(float* __restrict__ p, int n) {
  int i = blockIdx.x * blockDim.x + threadIdx.x;
  if (i < n) p[i] = 0.0f;
}

// msg = relu(h[src] + e), agg[dst] += msg ; one wave per edge, 4 channels per lane
__global__ void scatter_kernel(const float* __restrict__ h, const __bf16* __restrict__ e,
                               const int* __restrict__ ei, float* __restrict__ agg) {
  long tid = (long)blockIdx.x * blockDim.x + threadIdx.x;
  long edge = tid >> 5;
  if (edge >= NE) return;
  int c = ((int)tid & 31) * 4;
  int s = ei[edge], d = ei[NE + edge];
  float4 hv = *(const float4*)(h + (long)s * HD + c);
  v4bf ev = *(const v4bf*)(e + (long)edge * HD + c);
  float m0 = fmaxf(hv.x + (float)ev[0], 0.0f);
  float m1 = fmaxf(hv.y + (float)ev[1], 0.0f);
  float m2 = fmaxf(hv.z + (float)ev[2], 0.0f);
  float m3 = fmaxf(hv.w + (float)ev[3], 0.0f);
  float* a = agg + (long)d * HD + c;
  atomicAdd(a + 0, m0);
  atomicAdd(a + 1, m1);
  atomicAdd(a + 2, m2);
  atomicAdd(a + 3, m3);
}

__global__ void zbf_kernel(const float* __restrict__ h, const float* __restrict__ agg,
                           __bf16* __restrict__ zbf) {
  int i = blockIdx.x * blockDim.x + threadIdx.x;
  if (i < NN * HD) zbf[i] = (__bf16)(h[i] + agg[i]);
}

// per-channel sum / sumsq over all rows (register partials + f32 atomics)
__global__ void bn_stats_kernel(const float* __restrict__ y, float* __restrict__ stats) {
  int c = threadIdx.x;  // HD threads
  float s = 0.0f, s2 = 0.0f;
  for (int row = blockIdx.x; row < NN; row += gridDim.x) {
    float v = y[(long)row * HD + c];
    s += v; s2 += v * v;
  }
  atomicAdd(&stats[c], s);
  atomicAdd(&stats[HD + c], s2);
}

__global__ void bn_apply_kernel(const float* __restrict__ y, const float* __restrict__ stats,
                                const float* __restrict__ gamma, const float* __restrict__ beta,
                                float* __restrict__ h) {
  int i = blockIdx.x * blockDim.x + threadIdx.x;
  if (i >= NN * HD) return;
  int c = i & (HD - 1);
  const float invN = 1.0f / (float)NN;
  float mu  = stats[c] * invN;
  float var = stats[HD + c] * invN - mu * mu;
  float v = (y[i] - mu) * rsqrtf(var + BN_EPS) * gamma[c] + beta[c];
  h[i] = fmaxf(v, 0.0f);
}

__global__ void pool_kernel(const float* __restrict__ hf, const int* __restrict__ batch,
                            float* __restrict__ pooled, float* __restrict__ counts) {
  long tid = (long)blockIdx.x * blockDim.x + threadIdx.x;
  long node = tid >> 5;
  if (node >= NN) return;
  int c = ((int)tid & 31) * 4;
  int g = batch[node];
  float4 v = *(const float4*)(hf + node * HD + c);
  float* p = pooled + (long)g * HD + c;
  atomicAdd(p + 0, v.x);
  atomicAdd(p + 1, v.y);
  atomicAdd(p + 2, v.z);
  atomicAdd(p + 3, v.w);
  if (((int)tid & 31) == 0) atomicAdd(&counts[g], 1.0f);
}

__global__ void out_kernel(const float* __restrict__ pooled, const float* __restrict__ counts,
                           const float* __restrict__ ow, const float* __restrict__ ob,
                           float* __restrict__ out) {
  int i = blockIdx.x * blockDim.x + threadIdx.x;
  if (i >= NG * NC) return;
  int g = i / NC, c = i - g * NC;
  float inv = 1.0f / fmaxf(counts[g], 1.0f);
  const float* pr = pooled + (long)g * HD;
  float acc = ob[c];
#pragma unroll 8
  for (int k = 0; k < HD; ++k) acc += pr[k] * inv * ow[k * NC + c];
  out[i] = acc;
}

extern "C" void kernel_launch(void* const* d_in, const int* in_sizes, int n_in,
                              void* d_out, int out_size, void* d_ws, size_t ws_size,
                              hipStream_t stream) {
  const float* x      = (const float*)d_in[0];
  const float* eattr  = (const float*)d_in[1];
  const int*   ei     = (const int*)d_in[2];
  const int*   batch  = (const int*)d_in[3];
  const float* node_w = (const float*)d_in[4];
  const float* node_b = (const float*)d_in[5];
  const float* edge_w = (const float*)d_in[6];
  const float* edge_b = (const float*)d_in[7];
  const float* mw1    = (const float*)d_in[8];
  const float* mb1    = (const float*)d_in[9];
  const float* mw2    = (const float*)d_in[10];
  const float* mb2    = (const float*)d_in[11];
  const float* gamma  = (const float*)d_in[12];
  const float* beta   = (const float*)d_in[13];
  const float* fw     = (const float*)d_in[14];
  const float* fb     = (const float*)d_in[15];
  const float* ow     = (const float*)d_in[16];
  const float* ob     = (const float*)d_in[17];
  float* out = (float*)d_out;

  char* w = (char*)d_ws;
  size_t off = 0;
  auto carve = [&](size_t bytes) { void* p = w + off; off += (bytes + 255) & ~(size_t)255; return p; };
  float*  h      = (float*)carve((size_t)NN * HD * 4);
  float*  agg    = (float*)carve((size_t)NN * HD * 4);
  float*  y      = (float*)carve((size_t)NN * HD * 4);
  __bf16* e      = (__bf16*)carve((size_t)NE * HD * 2);
  __bf16* zbf    = (__bf16*)carve((size_t)NN * HD * 2);
  float*  stats  = (float*)carve(2 * HD * 4);
  float*  pooled = (float*)carve((size_t)NG * HD * 4);
  float*  counts = (float*)carve((size_t)NG * 4);
  (void)ws_size; (void)in_sizes; (void)n_in; (void)out_size;

  const int TPB = 128;                    // 4 wave32 per block, 64 rows per block
  int nbN = (NN + 63) / 64;
  int nbE = (NE + 63) / 64;

  gemm_kernel<FIN, 32, false, false><<<nbN, TPB, 0, stream>>>(x, node_w, node_b, (void*)h, NN);
  gemm_kernel<EDIM, 64, false, true><<<nbE, TPB, 0, stream>>>(eattr, edge_w, edge_b, (void*)e, NE);

  for (int l = 0; l < NL; ++l) {
    zero_kernel<<<(NN * HD + 255) / 256, 256, 0, stream>>>(agg, NN * HD);
    long sthreads = (long)NE * 32;
    scatter_kernel<<<(int)((sthreads + 255) / 256), 256, 0, stream>>>(h, e, ei, agg);
    zbf_kernel<<<(NN * HD + 255) / 256, 256, 0, stream>>>(h, agg, zbf);
    mlp_kernel<<<nbN, TPB, 0, stream>>>(zbf, mw1 + (size_t)l * HD * HD, mb1 + l * HD,
                                        mw2 + (size_t)l * HD * HD, mb2 + l * HD, y, NN);
    zero_kernel<<<1, 256, 0, stream>>>(stats, 2 * HD);
    bn_stats_kernel<<<512, HD, 0, stream>>>(y, stats);
    bn_apply_kernel<<<(NN * HD + 255) / 256, 256, 0, stream>>>(y, stats, gamma + l * HD,
                                                               beta + l * HD, h);
  }

  gemm_kernel<HD, HD, true, false><<<nbN, TPB, 0, stream>>>(h, fw, fb, (void*)y, NN);

  zero_kernel<<<(NG * HD + 255) / 256, 256, 0, stream>>>(pooled, NG * HD);
  zero_kernel<<<(NG + 255) / 256, 256, 0, stream>>>(counts, NG);
  long pthreads = (long)NN * 32;
  pool_kernel<<<(int)((pthreads + 255) / 256), 256, 0, stream>>>(y, batch, pooled, counts);
  out_kernel<<<(NG * NC + 255) / 256, 256, 0, stream>>>(pooled, counts, ow, ob, out);
}
